// MultiHeadSelfAttention_33569464386220
// MI455X (gfx1250) — compile-verified, matrix operand order checked by reading, not run on hardware
//
#include <hip/hip_runtime.h>
#include <hip/hip_bf16.h>
#include <math.h>

typedef __attribute__((ext_vector_type(16))) __bf16 v16bf;
typedef __attribute__((ext_vector_type(8)))  float  v8f;
typedef __attribute__((ext_vector_type(4)))  unsigned int u32x4;
typedef __attribute__((ext_vector_type(8)))  int i32x8;
typedef __attribute__((ext_vector_type(4)))  int i32x4;

#define DMODEL 1024
#define SEQ    2048
#define BATCH  2
#define NHEAD  16
#define DKH    64
#define MTOT   (BATCH * SEQ)   // 4096 rows

__device__ __forceinline__ int laneid() { return threadIdx.x & 31; }

// LDS byte offset (wave-relative) of a __shared__ pointer: addrspace(3)
// pointers on amdgcn are exactly the 32-bit LDS offsets the TDM D# wants.
__device__ __forceinline__ unsigned lds_offset(const void* p) {
  return (unsigned)(unsigned long long)(__attribute__((address_space(3))) const void*)p;
}

// Build a 16x32 bf16 WMMA fragment (A-layout per CDNA5 ISA 7.12.2) from a
// row-major bf16 LDS tile. B-fragments use the same layout with N as M.
__device__ __forceinline__ v16bf lds_frag(const __bf16* t, int row0, int stride, int col0) {
  const int lane = laneid();
  const __bf16* p = t + (row0 + (lane & 15)) * stride + col0 + ((lane >> 4) << 3);
  v16bf f;
#pragma unroll
  for (int j = 0; j < 8; ++j) {
    const int kb = (j < 4) ? (2 * j) : (8 + 2 * j);  // {0,2,4,6,16,18,20,22}
    f[2 * j]     = p[kb];
    f[2 * j + 1] = p[kb + 1];
  }
  return f;
}

// Same fragment build from an f32 LDS tile (TDM-staged), converting to bf16.
__device__ __forceinline__ v16bf lds_frag_f32(const float* t, int row0, int stride, int col0) {
  const int lane = laneid();
  const float* p = t + (row0 + (lane & 15)) * stride + col0 + ((lane >> 4) << 3);
  v16bf f;
#pragma unroll
  for (int j = 0; j < 8; ++j) {
    const int kb = (j < 4) ? (2 * j) : (8 + 2 * j);
    f[2 * j]     = (__bf16)p[kb];
    f[2 * j + 1] = (__bf16)p[kb + 1];
  }
  return f;
}

__device__ __forceinline__ v8f wmma_bf16(v16bf a, v16bf b, v8f c) {
  return __builtin_amdgcn_wmma_f32_16x16x32_bf16(false, a, false, b, (short)0, c,
                                                 false, false);
}

// ---- Tensor Data Mover: load a 128-row x 32-col f32 tile (row stride 1024
// elements in memory) into LDS with 2-dword padding per 32-dword row
// (LDS row stride = 34 floats). D# per CDNA5 ISA.
__device__ __forceinline__ i32x8 tdm_g1_128x32() {
  i32x8 g1 = {};
  // word0: data_size=4B (code 2) @[17:16], pad_enable @20,
  //        pad_interval=32 dwords (code 4) @[24:22], pad_amount=2 dwords (code 1) @[31:25]
  g1[0] = (2 << 16) | (1 << 20) | (4 << 22) | (1 << 25);
  g1[1] = (DMODEL & 0xFFFF) << 16;                        // tensor_dim0[15:0] @ bits63:48
  g1[2] = (DMODEL >> 16) | ((MTOT & 0xFFFF) << 16);       // tensor_dim0 hi | tensor_dim1 lo
  g1[3] = (MTOT >> 16) | (32 << 16);                      // tensor_dim1 hi | tile_dim0=32
  g1[4] = 128;                                            // tile_dim1=128 (tile_dim2=0)
  g1[5] = DMODEL;                                         // tensor_dim0_stride lo32
  g1[6] = 0;
  g1[7] = 0;
  return g1;
}

__device__ __forceinline__ void tdm_load(unsigned lds_off, unsigned long long gaddr,
                                         i32x8 g1) {
  u32x4 g0;
  g0[0] = 1u;                                             // count=1, user descriptor
  g0[1] = lds_off;                                        // lds_addr
  g0[2] = (unsigned)(gaddr & 0xFFFFFFFFu);                // global_addr[31:0]
  g0[3] = (unsigned)((gaddr >> 32) & 0x1FFFFFFu) | (2u << 30);  // addr[56:32] | type=2
  i32x4 z4 = {};
  i32x8 z8 = {};
  __builtin_amdgcn_tensor_load_to_lds(g0, g1, z4, z4, z8, 0);
}

// C = A[M x 1024] * W[1024 x 1024]^T  (einsum 'bsd,ed->bse')
// Block: 256 thr = 8 waves; 128x128 output tile; wave = 32(M) x 64(N).
// TDM double-buffers the 128x32 A/W tiles (f32) behind the WMMA work.
__global__ void __launch_bounds__(256) gemm_xwT(const float* __restrict__ A,
                                                const float* __restrict__ W,
                                                float* __restrict__ C) {
  __shared__ float sA[2][128 * 34];
  __shared__ float sB[2][128 * 34];
  const int m0 = blockIdx.x * 128;
  const int n0 = blockIdx.y * 128;
  const int t  = threadIdx.x;
  const int w  = t >> 5;
  const int wm = (w & 3) * 32;   // wave M offset within block tile
  const int wn = (w >> 2) * 64;  // wave N offset within block tile

  const unsigned long long Abase = (unsigned long long)A + (unsigned long long)m0 * DMODEL * 4ull;
  const unsigned long long Wbase = (unsigned long long)W + (unsigned long long)n0 * DMODEL * 4ull;
  const unsigned offA[2] = { lds_offset(&sA[0][0]), lds_offset(&sA[1][0]) };
  const unsigned offB[2] = { lds_offset(&sB[0][0]), lds_offset(&sB[1][0]) };
  const i32x8 g1 = tdm_g1_128x32();

  v8f acc[2][4];
  v8f vz = {};
#pragma unroll
  for (int mi = 0; mi < 2; ++mi)
#pragma unroll
    for (int ni = 0; ni < 4; ++ni) acc[mi][ni] = vz;

  if (w == 0) {
    tdm_load(offA[0], Abase, g1);
    tdm_load(offB[0], Wbase, g1);
  }

  int cur = 0;
  for (int kt = 0; kt < DMODEL; kt += 32) {
    if (w == 0) __builtin_amdgcn_s_wait_tensorcnt((short)0);
    __syncthreads();  // current tiles resident; previous compute finished
    if (w == 0 && kt + 32 < DMODEL) {
      tdm_load(offA[cur ^ 1], Abase + (unsigned long long)(kt + 32) * 4ull, g1);
      tdm_load(offB[cur ^ 1], Wbase + (unsigned long long)(kt + 32) * 4ull, g1);
    }
    const float* tA = sA[cur];
    const float* tB = sB[cur];
    v16bf fa0 = lds_frag_f32(tA, wm, 34, 0);
    v16bf fa1 = lds_frag_f32(tA, wm + 16, 34, 0);
#pragma unroll
    for (int ni = 0; ni < 4; ++ni) {
      v16bf fb = lds_frag_f32(tB, wn + ni * 16, 34, 0);
      acc[0][ni] = wmma_bf16(fa0, fb, acc[0][ni]);
      acc[1][ni] = wmma_bf16(fa1, fb, acc[1][ni]);
    }
    cur ^= 1;
  }

  const int lane  = laneid();
  const int nbase = n0 + wn + (lane & 15);
  const int mbase = m0 + wm + ((lane >> 4) << 3);
#pragma unroll
  for (int mi = 0; mi < 2; ++mi)
#pragma unroll
    for (int j = 0; j < 8; ++j) {
      const int row = mbase + mi * 16 + j;
#pragma unroll
      for (int ni = 0; ni < 4; ++ni)
        C[(size_t)row * DMODEL + nbase + ni * 16] = acc[mi][ni][j];
    }
}

// In-place RoPE on q and k. One thread per (b,s,h,i) rotation pair.
__global__ void __launch_bounds__(256) rope_qk(float* __restrict__ q,
                                               float* __restrict__ k,
                                               const int* __restrict__ pos) {
  int idx = blockIdx.x * blockDim.x + threadIdx.x;  // 2^21 total
  int i = idx & 31;
  int h = (idx >> 5) & 15;
  int s = (idx >> 9) & (SEQ - 1);
  int b = idx >> 20;
  float p   = (float)pos[s];
  float inv = __powf(10000.0f, -(float)(2 * i) / 64.0f);
  float sn, cs;
  __sincosf(p * inv, &sn, &cs);
  size_t base = (size_t)(b * SEQ + s) * DMODEL + h * DKH + 2 * i;
  float q1 = q[base], q2 = q[base + 1];
  q[base]     = q1 * cs - q2 * sn;
  q[base + 1] = q1 * sn + q2 * cs;
  float k1 = k[base], k2 = k[base + 1];
  k[base]     = k1 * cs - k2 * sn;
  k[base + 1] = k1 * sn + k2 * cs;
}

// Flash attention: block = (b,h,qtile of 64 rows), 128 thr = 4 waves,
// each wave owns a 16-row Q block. KV streamed in 32-row tiles through LDS.
__global__ void __launch_bounds__(128) flash_attn(const float* __restrict__ q,
                                                  const float* __restrict__ k,
                                                  const float* __restrict__ v,
                                                  float* __restrict__ o) {
  __shared__ __bf16 sQ[64 * 66];       // 64 x 64 (dk), padded
  __shared__ __bf16 sK[32 * 66];       // 32 kv rows x 64 dk
  __shared__ __bf16 sVt[64 * 34];      // V transposed: [dk][kv]
  __shared__ __bf16 sP[4 * 16 * 34];   // per-wave 16x32 prob tile
  const int bh = blockIdx.x;
  const int b  = bh >> 4, h = bh & 15;
  const int qt = blockIdx.y;
  const int t  = threadIdx.x;
  const int w  = t >> 5;
  const int lane = t & 31;
  const int qrow0 = b * SEQ + qt * 64;
  const int hoff  = h * DKH;

#pragma unroll
  for (int i = 0; i < 32; ++i) {
    int idx = t + i * 128;
    int r = idx >> 6, c = idx & 63;
    sQ[r * 66 + c] = (__bf16)q[(size_t)(qrow0 + r) * DMODEL + hoff + c];
  }
  __syncthreads();
  v16bf qa0 = lds_frag(sQ, w * 16, 66, 0);
  v16bf qa1 = lds_frag(sQ, w * 16, 66, 32);

  v8f oacc[4];
  float mrow[8], lsum[8];
  v8f vz = {};
#pragma unroll
  for (int dt = 0; dt < 4; ++dt) oacc[dt] = vz;
#pragma unroll
  for (int j = 0; j < 8; ++j) { mrow[j] = -INFINITY; lsum[j] = 0.f; }

  const int mglob = qt * 64 + w * 16 + ((lane >> 4) << 3);  // + j -> q index
  const int ncol  = lane & 15;
  const int ntiles = qt * 2 + 2;  // causal: kv tiles with kv0 <= qtile end
  for (int nt = 0; nt < ntiles; ++nt) {
    const int kv0 = nt * 32;
    __syncthreads();
#pragma unroll
    for (int i = 0; i < 16; ++i) {
      int idx = t + i * 128;
      int r = idx >> 6, c = idx & 63;
      size_t g = (size_t)(b * SEQ + kv0 + r) * DMODEL + hoff + c;
      sK[r * 66 + c]  = (__bf16)k[g];
      sVt[c * 34 + r] = (__bf16)v[g];
    }
    __syncthreads();

    v16bf kb00 = lds_frag(sK, 0, 66, 0);
    v16bf kb01 = lds_frag(sK, 0, 66, 32);
    v16bf kb10 = lds_frag(sK, 16, 66, 0);
    v16bf kb11 = lds_frag(sK, 16, 66, 32);
    v8f s0 = {}, s1 = {};
    s0 = wmma_bf16(qa0, kb00, s0);
    s0 = wmma_bf16(qa1, kb01, s0);
    s1 = wmma_bf16(qa0, kb10, s1);
    s1 = wmma_bf16(qa1, kb11, s1);

    const float scale = 0.125f;  // 1/sqrt(64)
    float p0[8], p1[8];
#pragma unroll
    for (int j = 0; j < 8; ++j) {
      int mg = mglob + j;
      float a0 = (kv0 + ncol      <= mg) ? s0[j] * scale : -INFINITY;
      float a1 = (kv0 + 16 + ncol <= mg) ? s1[j] * scale : -INFINITY;
      float tm = fmaxf(a0, a1);
#pragma unroll
      for (int msk = 1; msk < 16; msk <<= 1) tm = fmaxf(tm, __shfl_xor(tm, msk, 32));
      float nm = fmaxf(fmaxf(mrow[j], tm), -1e30f);
      float alpha = __expf(mrow[j] - nm);
      float e0 = __expf(a0 - nm), e1 = __expf(a1 - nm);
      float rs = e0 + e1;
#pragma unroll
      for (int msk = 1; msk < 16; msk <<= 1) rs += __shfl_xor(rs, msk, 32);
      lsum[j] = lsum[j] * alpha + rs;
      mrow[j] = nm;
#pragma unroll
      for (int dt = 0; dt < 4; ++dt) oacc[dt][j] *= alpha;
      p0[j] = e0; p1[j] = e1;
    }

    // C-layout -> A-layout through per-wave LDS (same-wave RAW: wait dscnt)
    __bf16* pw = sP + w * (16 * 34);
#pragma unroll
    for (int j = 0; j < 8; ++j) {
      int m = j + ((lane >> 4) << 3);
      pw[m * 34 + ncol]      = (__bf16)p0[j];
      pw[m * 34 + ncol + 16] = (__bf16)p1[j];
    }
    asm volatile("s_wait_dscnt 0" ::: "memory");
    v16bf pf = lds_frag(pw, 0, 34, 0);
#pragma unroll
    for (int dt = 0; dt < 4; ++dt) {
      v16bf vb = lds_frag(sVt, dt * 16, 34, 0);
      oacc[dt] = wmma_bf16(pf, vb, oacc[dt]);
    }
  }

#pragma unroll
  for (int j = 0; j < 8; ++j) {
    int m = j + ((lane >> 4) << 3);
    int grow = qrow0 + w * 16 + m;
    float invl = 1.0f / lsum[j];
#pragma unroll
    for (int dt = 0; dt < 4; ++dt)
      o[(size_t)grow * DMODEL + hoff + dt * 16 + ncol] = oacc[dt][j] * invl;
  }
}

extern "C" void kernel_launch(void* const* d_in, const int* in_sizes, int n_in,
                              void* d_out, int out_size, void* d_ws, size_t ws_size,
                              hipStream_t stream) {
  const float* x  = (const float*)d_in[0];
  const int*  pos = (const int*)d_in[1];
  const float* wq = (const float*)d_in[2];
  const float* wk = (const float*)d_in[3];
  const float* wv = (const float*)d_in[4];
  const float* wo = (const float*)d_in[5];
  float* out = (float*)d_out;

  float* ws = (float*)d_ws;
  float* q  = ws;
  float* kk = ws + (size_t)1 * MTOT * DMODEL;
  float* vv = ws + (size_t)2 * MTOT * DMODEL;
  float* ao = ws + (size_t)3 * MTOT * DMODEL;

  dim3 gg(MTOT / 128, DMODEL / 128);
  gemm_xwT<<<gg, 256, 0, stream>>>(x, wq, q);
  gemm_xwT<<<gg, 256, 0, stream>>>(x, wk, kk);
  gemm_xwT<<<gg, 256, 0, stream>>>(x, wv, vv);
  rope_qk<<<(BATCH * SEQ * NHEAD * 32) / 256, 256, 0, stream>>>(q, kk, pos);
  flash_attn<<<dim3(BATCH * NHEAD, SEQ / 64), 128, 0, stream>>>(q, kk, vv, ao);
  gemm_xwT<<<gg, 256, 0, stream>>>(ao, wo, out);
}
